// SelfAttention_19945828123228
// MI455X (gfx1250) — matrix operand order repaired and checked by reading, not compiled, not run on hardware
//
#include <hip/hip_runtime.h>

// Problem constants (B, C, N) from the reference.
#define BB 8
#define CC 512
#define NN 2048

#define SCHED_FENCE() __builtin_amdgcn_sched_barrier(0)

typedef __attribute__((ext_vector_type(16))) __bf16       v16bf;
typedef __attribute__((ext_vector_type(8)))  float        v8f;
typedef __attribute__((ext_vector_type(8)))  unsigned int v8u;

static __device__ inline unsigned short f2bfbits(float f) {
  unsigned u = __builtin_bit_cast(unsigned, f);
  u += 0x7FFFu + ((u >> 16) & 1u);               // round-to-nearest-even
  return (unsigned short)(u >> 16);
}

static __device__ inline unsigned pk2bf(float a, float b) {
  return (unsigned)f2bfbits(a) | ((unsigned)f2bfbits(b) << 16);
}

static __device__ inline v8u pack8(uint4 lo, uint4 hi) {
  v8u r;
  r[0] = lo.x; r[1] = lo.y; r[2] = lo.z; r[3] = lo.w;
  r[4] = hi.x; r[5] = hi.y; r[6] = hi.z; r[7] = hi.w;
  return r;
}

// B-fragment (32x16 bf16): chunks at +0..7 and +16..23 (lane pre-offset by koff).
static __device__ inline v8u loadB16(const unsigned short* p) {
  uint4 lo = *(const uint4*)(p);
  uint4 hh = *(const uint4*)(p + 16);
  return pack8(lo, hh);
}
// A-fragment (16x32 bf16): chunks at +0..7 and +16..23 (lane pre-offset by koff).
static __device__ inline v8u loadA32(const unsigned short* p) {
  uint4 lo = *(const uint4*)(p);
  uint4 hh = *(const uint4*)(p + 16);
  return pack8(lo, hh);
}

static __device__ inline v8f wmma_bf16(v8u a, v8u b, v8f c) {
  return __builtin_amdgcn_wmma_f32_16x16x32_bf16(
      false, __builtin_bit_cast(v16bf, a),
      false, __builtin_bit_cast(v16bf, b),
      (short)0, c, false, false);
}

// ---------------------------------------------------------------------------
// Kernel 0: cast Wq|Wk|Wv (fp32, [d][c]) to bf16 into workspace (vectorized).
// ---------------------------------------------------------------------------
__global__ __launch_bounds__(256) void conv_w_kernel(
    const float* __restrict__ Wq, const float* __restrict__ Wk,
    const float* __restrict__ Wv, unsigned short* __restrict__ wbf) {
  const int per4 = CC * CC / 4;
  int i = blockIdx.x * 256 + threadIdx.x;         // group of 4 elements
  const float* src;
  if (i < per4)            src = Wq + (size_t)i * 4;
  else if (i < 2 * per4)   src = Wk + (size_t)(i - per4) * 4;
  else                     src = Wv + (size_t)(i - 2 * per4) * 4;
  float4 v = *(const float4*)src;
  uint2 o;
  o.x = pk2bf(v.x, v.y);
  o.y = pk2bf(v.z, v.w);
  ((uint2*)wbf)[i] = o;
}

// ---------------------------------------------------------------------------
// Kernel 1: fused QKV projection.
// Block = (b, n0): 16 query rows. Stages x^T tile [16][512] in LDS as bf16,
// each of 8 waves computes 12 of the 96 16x16 output tiles (3 mats x 32 dtiles),
// reusing its preloaded A-fragments. B-fragments are double-buffered in
// groups of 4 K-steps; sched_barrier pins the prefetch ahead of the WMMAs.
// Q,K stored [b][n][d] bf16; V stored transposed [b][d][n] bf16.
// ---------------------------------------------------------------------------
__global__ __launch_bounds__(256) void qkv_kernel(
    const float* __restrict__ x, const unsigned short* __restrict__ wbf,
    const float* __restrict__ bq, const float* __restrict__ bk,
    const float* __restrict__ bv,
    unsigned short* __restrict__ qb, unsigned short* __restrict__ kb,
    unsigned short* __restrict__ vtb) {
  __shared__ __align__(16) unsigned short lds_x[16 * CC];

  const int b  = blockIdx.y;
  const int n0 = blockIdx.x * 16;
  const float* xb = x + (size_t)b * CC * NN;      // x is [C][N] per batch

  // Stage x^T tile: float4 coalesced global loads, scalar bf16 LDS stores.
  for (int t = threadIdx.x; t < 16 * CC / 4; t += 256) {
    int jj = t & 3, c = t >> 2;
    float4 vv = *(const float4*)(xb + (size_t)c * NN + n0 + jj * 4);
    int i = jj * 4;
    lds_x[(i + 0) * CC + c] = f2bfbits(vv.x);
    lds_x[(i + 1) * CC + c] = f2bfbits(vv.y);
    lds_x[(i + 2) * CC + c] = f2bfbits(vv.z);
    lds_x[(i + 3) * CC + c] = f2bfbits(vv.w);
  }
  __syncthreads();

  const int wave = threadIdx.x >> 5;
  const int lane = threadIdx.x & 31;
  const int row  = lane & 15;
  const int hi16 = lane >> 4;
  const int mb   = hi16 * 8;

  // Preload all 16 K-step A-fragments (shared across this wave's 12 tiles).
  v8u afrag[16];
  {
    const unsigned short* base = &lds_x[row * CC + (hi16 ? 8 : 0)];
#pragma unroll
    for (int s = 0; s < 16; ++s) afrag[s] = loadA32(base + s * 32);
  }

  for (int t = wave; t < 96; t += 8) {
    const int mat = t >> 5;                       // 0=Q 1=K 2=V
    const int d0  = (t & 31) * 16;
    const unsigned short* wrow =
        wbf + (size_t)mat * CC * CC + (size_t)(d0 + row) * CC + (hi16 ? 8 : 0);

    v8u bbuf[2][4];
#pragma unroll
    for (int i = 0; i < 4; ++i) bbuf[0][i] = loadB16(wrow + i * 32);
    v8f acc = {};
#pragma unroll
    for (int g = 0; g < 4; ++g) {
      if (g < 3) {
#pragma unroll
        for (int i = 0; i < 4; ++i)
          bbuf[(g + 1) & 1][i] = loadB16(wrow + ((g + 1) * 4 + i) * 32);
      }
      SCHED_FENCE();   // keep next-group loads ahead of this group's WMMAs
#pragma unroll
      for (int i = 0; i < 4; ++i)
        acc = wmma_bf16(afrag[g * 4 + i], bbuf[g & 1][i], acc);
      SCHED_FENCE();
    }

    const float* bias = (mat == 0) ? bq : (mat == 1) ? bk : bv;
    const float bb = bias[d0 + row];
    if (mat == 2) {                               // V -> transposed [d][n]
      unsigned short* dst = vtb + ((size_t)b * CC + d0 + row) * NN + n0 + mb;
      uint4 o;
      o.x = pk2bf(acc[0] + bb, acc[1] + bb);
      o.y = pk2bf(acc[2] + bb, acc[3] + bb);
      o.z = pk2bf(acc[4] + bb, acc[5] + bb);
      o.w = pk2bf(acc[6] + bb, acc[7] + bb);
      *(uint4*)dst = o;
    } else {                                      // Q/K -> [n][d]
      unsigned short* dst =
          (mat == 0 ? qb : kb) + ((size_t)b * NN + n0 + mb) * CC + d0 + row;
#pragma unroll
      for (int r = 0; r < 8; ++r) dst[(size_t)r * CC] = f2bfbits(acc[r] + bb);
    }
  }
}

// ---------------------------------------------------------------------------
// Kernel 2: attention for one 16-row query block.
// S[16][2048] fp32 lives entirely in LDS (128 KB of the 320 KB WGP LDS);
// exact softmax in LDS; P bf16 in LDS feeds P*V WMMAs via ds_load_b128.
// All WMMA loops are double-buffered with sched_barrier fences;
// epilogue fuses transpose + residual.
// ---------------------------------------------------------------------------
__global__ __launch_bounds__(256) void attn_kernel(
    const float* __restrict__ x,
    const unsigned short* __restrict__ qb, const unsigned short* __restrict__ kb,
    const unsigned short* __restrict__ vtb, float* __restrict__ out) {
  extern __shared__ char smem_raw[];
  float*          S   = (float*)smem_raw;                                // 16*2048 f32
  unsigned short* P   = (unsigned short*)(smem_raw + 16 * NN * 4);       // 16*2048 bf16
  float*          red = (float*)(smem_raw + 16 * NN * 4 + 16 * NN * 2);  // 256 f32

  const int b    = blockIdx.y;
  const int n0   = blockIdx.x * 16;
  const int wave = threadIdx.x >> 5;
  const int lane = threadIdx.x & 31;
  const int row  = lane & 15;
  const int hi16 = lane >> 4;
  const int mb   = hi16 * 8;

  // Preload all Q A-fragments for this row block.
  v8u qfrag[16];
  {
    const unsigned short* qrow =
        qb + ((size_t)b * NN + n0 + row) * CC + (hi16 ? 8 : 0);
#pragma unroll
    for (int s = 0; s < 16; ++s) qfrag[s] = loadA32(qrow + s * 32);
  }

  // ---- Phase 1: S = Q K^T (128 m-tiles, 16 per wave, pipelined B loads) ----
  for (int j = wave; j < NN / 16; j += 8) {
    const int m0 = j * 16;
    const unsigned short* krow =
        kb + ((size_t)b * NN + m0 + row) * CC + (hi16 ? 8 : 0);

    v8u bbuf[2][4];
#pragma unroll
    for (int i = 0; i < 4; ++i) bbuf[0][i] = loadB16(krow + i * 32);
    v8f acc = {};
#pragma unroll
    for (int g = 0; g < 4; ++g) {
      if (g < 3) {
#pragma unroll
        for (int i = 0; i < 4; ++i)
          bbuf[(g + 1) & 1][i] = loadB16(krow + ((g + 1) * 4 + i) * 32);
      }
      SCHED_FENCE();
#pragma unroll
      for (int i = 0; i < 4; ++i)
        acc = wmma_bf16(qfrag[g * 4 + i], bbuf[g & 1][i], acc);
      SCHED_FENCE();
    }

    float* sp = S + (size_t)mb * NN + m0 + row;   // (M, N=m0+row)
#pragma unroll
    for (int r = 0; r < 8; ++r) sp[(size_t)r * NN] = acc[r];
  }
  __syncthreads();

  // ---- Phase 2: exact softmax over each of the 16 rows (vectorized) ----
  {
    const int srow = threadIdx.x >> 4;            // 0..15
    const int seg  = threadIdx.x & 15;            // 0..15, 128 cols each
    float4* rp4 = (float4*)(S + srow * NN + seg * 128);
    float mx = -3.4e38f;
    for (int c2 = 0; c2 < 32; ++c2) {
      float4 vv = rp4[c2];
      mx = fmaxf(mx, fmaxf(fmaxf(vv.x, vv.y), fmaxf(vv.z, vv.w)));
    }
    red[srow * 16 + seg] = mx;
    __syncthreads();
    float m = -3.4e38f;
    for (int i = 0; i < 16; ++i) m = fmaxf(m, red[srow * 16 + i]);
    __syncthreads();
    // softmax of S/sqrt(C): exp2((S-max)*inv_scale*log2e).
    const float sc = 0.044194173824159216f * 1.4426950408889634f;
    float ssum = 0.f;
    for (int c2 = 0; c2 < 32; ++c2) {
      float4 vv = rp4[c2];
      vv.x = exp2f((vv.x - m) * sc);
      vv.y = exp2f((vv.y - m) * sc);
      vv.z = exp2f((vv.z - m) * sc);
      vv.w = exp2f((vv.w - m) * sc);
      rp4[c2] = vv;
      ssum += (vv.x + vv.y) + (vv.z + vv.w);
    }
    red[srow * 16 + seg] = ssum;
    __syncthreads();
    float tot = 0.f;
    for (int i = 0; i < 16; ++i) tot += red[srow * 16 + i];
    const float inv = 1.f / tot;
    uint2* pp2 = (uint2*)(P + srow * NN + seg * 128);
    for (int c2 = 0; c2 < 32; ++c2) {
      float4 vv = rp4[c2];
      uint2 o;
      o.x = pk2bf(vv.x * inv, vv.y * inv);
      o.y = pk2bf(vv.z * inv, vv.w * inv);
      pp2[c2] = o;
    }
  }
  __syncthreads();

  // ---- Phase 3: O = P V (A from LDS, B = V^T rows; both double-buffered) ----
  v8f oacc[4] = {};
  const unsigned short* prow = P + row * NN + (hi16 ? 8 : 0);
  const unsigned short* vtbB =
      vtb + (size_t)b * CC * NN + (size_t)row * NN + (hi16 ? 8 : 0);

  v8u a0, a1, b0[4], b1[4];
  a0 = loadA32(prow);
#pragma unroll
  for (int t = 0; t < 4; ++t)
    b0[t] = loadB16(vtbB + (size_t)((wave + t * 8) * 16) * NN);

  for (int s = 0; s < NN / 32; s += 2) {
    // fill odd buffers (s+1 <= 63 always)
    a1 = loadA32(prow + (s + 1) * 32);
#pragma unroll
    for (int t = 0; t < 4; ++t)
      b1[t] = loadB16(vtbB + (size_t)((wave + t * 8) * 16) * NN + (s + 1) * 32);
    SCHED_FENCE();   // pin odd-buffer loads ahead of even-buffer WMMAs
#pragma unroll
    for (int t = 0; t < 4; ++t) oacc[t] = wmma_bf16(a0, b0[t], oacc[t]);
    SCHED_FENCE();

    if (s + 2 < NN / 32) {
      a0 = loadA32(prow + (s + 2) * 32);
#pragma unroll
      for (int t = 0; t < 4; ++t)
        b0[t] = loadB16(vtbB + (size_t)((wave + t * 8) * 16) * NN + (s + 2) * 32);
    }
    SCHED_FENCE();   // pin even-buffer loads ahead of odd-buffer WMMAs
#pragma unroll
    for (int t = 0; t < 4; ++t) oacc[t] = wmma_bf16(a1, b1[t], oacc[t]);
    SCHED_FENCE();
  }

  // ---- Epilogue: out[b][c][n] = O[n][c] + x[b][c][n] (float4-packed) ----
#pragma unroll
  for (int t = 0; t < 4; ++t) {
    const int d0 = (wave + t * 8) * 16;
    const size_t base = ((size_t)b * CC + d0 + row) * NN + n0 + mb;
    float4 x0 = *(const float4*)(x + base);
    float4 x1 = *(const float4*)(x + base + 4);
    float4 o0, o1;
    o0.x = oacc[t][0] + x0.x; o0.y = oacc[t][1] + x0.y;
    o0.z = oacc[t][2] + x0.z; o0.w = oacc[t][3] + x0.w;
    o1.x = oacc[t][4] + x1.x; o1.y = oacc[t][5] + x1.y;
    o1.z = oacc[t][6] + x1.z; o1.w = oacc[t][7] + x1.w;
    *(float4*)(out + base)     = o0;
    *(float4*)(out + base + 4) = o1;
  }
}

// ---------------------------------------------------------------------------
extern "C" void kernel_launch(void* const* d_in, const int* in_sizes, int n_in,
                              void* d_out, int out_size, void* d_ws, size_t ws_size,
                              hipStream_t stream) {
  (void)in_sizes; (void)n_in; (void)out_size; (void)ws_size;
  const float* x  = (const float*)d_in[0];
  const float* Wq = (const float*)d_in[1];
  const float* bq = (const float*)d_in[2];
  const float* Wk = (const float*)d_in[3];
  const float* bk = (const float*)d_in[4];
  const float* Wv = (const float*)d_in[5];
  const float* bv = (const float*)d_in[6];
  float* out = (float*)d_out;

  // Workspace layout (bf16 elements): weights | Q | K | V^T  (~49.5 MB)
  unsigned short* ws  = (unsigned short*)d_ws;
  unsigned short* wbf = ws;
  unsigned short* qb  = ws + (size_t)3 * CC * CC;
  unsigned short* kb  = qb + (size_t)BB * NN * CC;
  unsigned short* vtb = kb + (size_t)BB * NN * CC;

  conv_w_kernel<<<(3 * CC * CC / 4) / 256, 256, 0, stream>>>(Wq, Wk, Wv, wbf);
  qkv_kernel<<<dim3(NN / 16, BB), 256, 0, stream>>>(x, wbf, bq, bk, bv, qb, kb, vtb);
  const size_t smem = (size_t)16 * NN * 4 + (size_t)16 * NN * 2 + 1024;
  attn_kernel<<<dim3(NN / 16, BB), 256, smem, stream>>>(x, qb, kb, vtb, out);
}
